// CConvDecoder_34041910788793
// MI455X (gfx1250) — compile-verified, matrix operand order checked by reading, not run on hardware
//
#include <hip/hip_runtime.h>

typedef __attribute__((ext_vector_type(16))) _Float16 v16h;
typedef __attribute__((ext_vector_type(8)))  _Float16 v8h;
typedef __attribute__((ext_vector_type(8)))  float    v8f;

#define GRIDN   64
#define DXF     (1.0f/64.0f)
#define RADIUSF (2.5f/64.0f)
#define R2F     (RADIUSF*RADIUSF)
#define INV_R   (1.0f/RADIUSF)
#define EPSF    1e-12f
#define FOPI    1.2732395447351628f   // 4/pi
#define CH      16
#define CSTRIDE 262144                // 64^3 channel stride in floats
#define KACC    224                   // candidates padded: 7 * 32
#define KVAL    432                   // 27 taps * 16 ch
#define KF      512                   // final K padded: 32 taps * 16 ch
#define PTS     16                    // points per block (1 wave)

__device__ __forceinline__ float ssqrtf_(float v) { return __builtin_sqrtf(fmaxf(v, EPSF)); }
__device__ __forceinline__ float sgnf_(float v)   { return (v > 0.f) ? 1.f : ((v < 0.f) ? -1.f : 0.f); }

// volume-preserving ball -> cylinder -> cube map (matches reference)
__device__ __forceinline__ void map_ball_to_cube(float x, float y, float z,
                                                 float& X, float& Y, float& Z) {
    float sq  = x*x + y*y + z*z;
    float nrm = ssqrtf_(sq);
    float xy2 = x*x + y*y;
    bool  cap = 1.25f * z * z > xy2;
    float s_cap  = __builtin_sqrtf(3.f * nrm / (nrm + fabsf(z) + EPSF));
    float s_side = nrm / ssqrtf_(xy2);
    float xc = cap ? x * s_cap : x * s_side;
    float yc = cap ? y * s_cap : y * s_side;
    float zc = cap ? sgnf_(z) * nrm : 1.5f * z;
    bool zer = sq <= EPSF;
    x = zer ? 0.f : xc;  y = zer ? 0.f : yc;  z = zer ? 0.f : zc;

    float xy2b = x*x + y*y;
    float nxy  = ssqrtf_(xy2b);
    bool  xdom = fabsf(y) <= fabsf(x);
    float safex = (fabsf(x) > EPSF) ? x : 1.f;
    float safey = (fabsf(y) > EPSF) ? y : 1.f;
    float tx = sgnf_(x) * nxy;
    float ty = sgnf_(y) * nxy;
    float xa = tx;
    float ya = tx * FOPI * atanf(y / safex);
    float xb = ty * FOPI * atanf(x / safey);
    float yb = ty;
    float xr = xdom ? xa : xb;
    float yr = xdom ? ya : yb;
    bool zer2 = xy2b <= EPSF;
    X = zer2 ? 0.f : xr;
    Y = zer2 ? 0.f : yr;
    Z = z;
}

__device__ __forceinline__ float hatw(float v, float tap) {
    float c = fminf(fmaxf(v + 1.f, 0.f), 2.f);
    return fmaxf(0.f, 1.f - fabsf(c - tap));
}

__device__ __forceinline__ v8f wmma_f16(const v16h& a, const v16h& b, const v8f& c) {
    return __builtin_amdgcn_wmma_f32_16x16x32_f16(false, a, false, b, (short)0, c, false, false);
}

__device__ __forceinline__ v16h pack16(const v8h lo, const v8h hi) {
    v16h r;
    #pragma unroll
    for (int e = 0; e < 8; ++e) { r[e] = lo[e]; r[e + 8] = hi[e]; }
    return r;
}

__global__ __launch_bounds__(32)
void cconv_decoder_kernel(const float* __restrict__ input,     // [2,16,64,64,64]
                          const float* __restrict__ pos,       // [2,8192,3]
                          const float* __restrict__ weights,   // [27,16,16] flat
                          float* __restrict__ out)             // [2,8192,16]
{
    // A-tile of the accumulation GEMM: At[tap][cand] = w27[cand, tap]  (rows 27..31 stay 0)
    __shared__ __align__(16) _Float16 At[32][KACC];     // 14336 B
    // B-tile (transposed storage): Bfg[ch][cand] = feat[cand, ch]
    __shared__ __align__(16) _Float16 Bfg[CH][KACC];    //  7168 B
    // Final GEMM A-tile: Afin[point][tap*16+ch]
    __shared__ __align__(16) _Float16 Afin[PTS][KF];    // 16384 B
    // Final GEMM B-tile (transposed): Bt[d][k] = W[k, d]
    __shared__ __align__(16) _Float16 Bt[CH][KF];       // 16384 B
    __shared__ float cntS[PTS];

    const int lane = threadIdx.x;          // 0..31, one wave
    __builtin_prefetch(weights, 0, 3);

    // Build Bt (weights, transposed, zero-padded K)
    for (int idx = lane; idx < CH * KF; idx += 32) {
        int d = idx / KF, k = idx % KF;
        Bt[d][k] = (_Float16)((k < KVAL) ? weights[k * 16 + d] : 0.f);
    }
    // Zero unused tap rows 27..31 of At (written once; never touched again)
    for (int idx = lane; idx < 5 * KACC; idx += 32) {
        At[27 + idx / KACC][idx % KACC] = (_Float16)0.f;
    }
    __syncthreads();

    const int tileBase = blockIdx.x * PTS;
    const int Mrow  = lane & 15;
    const int half8 = (lane >> 4) * 8;

    for (int p = 0; p < PTS; ++p) {
        const int ng = tileBase + p;
        const int b  = ng >> 13;
        const float* pp = pos + (size_t)ng * 3;
        const float px = pp[0], py = pp[1], pz = pp[2];
        const int bx = (int)floorf(px * (float)GRIDN);
        const int by = (int)floorf(py * (float)GRIDN);
        const int bz = (int)floorf(pz * (float)GRIDN);
        const float* fbase = input + (size_t)b * CH * CSTRIDE;
        int count = 0;

        // ---- build tiles: lane = candidate, 7 iterations of 32 ----
        #pragma unroll 1
        for (int it = 0; it < 7; ++it) {
            const int k  = it * 32 + lane;          // tile column
            const int kk = min(k, 215);             // clamp pads to a real cell
            const int o0 = kk / 36 - 2, o1 = (kk / 6) % 6 - 2, o2 = kk % 6 - 2;
            const int c0 = bx + o0, c1 = by + o1, c2 = bz + o2;
            const bool inb = ((unsigned)c0 < GRIDN) && ((unsigned)c1 < GRIDN) && ((unsigned)c2 < GRIDN);
            const int q0 = min(max(c0, 0), GRIDN - 1);
            const int q1 = min(max(c1, 0), GRIDN - 1);
            const int q2 = min(max(c2, 0), GRIDN - 1);
            const int flat = (q0 * GRIDN + q1) * GRIDN + q2;
            const float rx = q0 * DXF - px;
            const float ry = q1 * DXF - py;
            const float rz = q2 * DXF - pz;
            const float d2 = rx * rx + ry * ry + rz * rz;
            const bool valid = inb && (d2 <= R2F) && (k < 216);
            count += (int)__popcll(__ballot(valid));

            float X, Y, Z;
            map_ball_to_cube(rx * INV_R, ry * INV_R, rz * INV_R, X, Y, Z);
            const float vf = valid ? 1.f : 0.f;
            float hx[3], hy[3], hz[3];
            #pragma unroll
            for (int s = 0; s < 3; ++s) {
                hx[s] = hatw(X, (float)s);
                hy[s] = hatw(Y, (float)s);
                hz[s] = hatw(Z, (float)s) * vf;     // fold validity mask into one factor
            }
            // 27 tap weights -> At[tap][k]
            #pragma unroll
            for (int i3 = 0; i3 < 3; ++i3)
                #pragma unroll
                for (int j3 = 0; j3 < 3; ++j3) {
                    const float hxy = hx[i3] * hy[j3];
                    #pragma unroll
                    for (int l3 = 0; l3 < 3; ++l3)
                        At[(i3 * 3 + j3) * 3 + l3][k] = (_Float16)(hxy * hz[l3]);
                }
            // 16 channels of this cell -> Bfg[ch][k] (pads store finite garbage; A col is 0)
            #pragma unroll
            for (int c = 0; c < CH; ++c)
                Bfg[c][k] = (_Float16)fbase[(size_t)c * CSTRIDE + flat];
        }
        __syncthreads();   // single-wave WG: fence + dscnt wait only

        // ---- accumulation GEMM: acc[32 taps x 16 ch] = At(32x224) x Bfg^T(224x16) ----
        v8f d0 = {0.f,0.f,0.f,0.f,0.f,0.f,0.f,0.f};
        v8f d1 = {0.f,0.f,0.f,0.f,0.f,0.f,0.f,0.f};
        #pragma unroll
        for (int q = 0; q < 7; ++q) {
            const int kb = q * 32;
            v16h bv = pack16(*(const v8h*)&Bfg[Mrow][kb + half8],
                             *(const v8h*)&Bfg[Mrow][kb + 16 + half8]);
            v16h a0 = pack16(*(const v8h*)&At[Mrow][kb + half8],
                             *(const v8h*)&At[Mrow][kb + 16 + half8]);
            v16h a1 = pack16(*(const v8h*)&At[16 + Mrow][kb + half8],
                             *(const v8h*)&At[16 + Mrow][kb + 16 + half8]);
            d0 = wmma_f16(a0, bv, d0);
            d1 = wmma_f16(a1, bv, d1);
        }

        // ---- scatter acc (f32->f16) into final A-tile row p ----
        // D layout: VGPR r -> tap = 16*blk + r + half8, ch = lane&15
        #pragma unroll
        for (int r = 0; r < 8; ++r) {
            const int tap0 = r + half8;           // taps 0..15
            const int tap1 = 16 + r + half8;      // taps 16..31 (27..31 are exact zeros)
            Afin[p][tap0 * 16 + Mrow] = (_Float16)d0[r];
            Afin[p][tap1 * 16 + Mrow] = (_Float16)d1[r];
        }
        if (lane == 0) cntS[p] = (float)count;
        __syncthreads();
    }

    __syncthreads();

    // ---- final GEMM: out[16 pts x 16 ch] = Afin(16x512) x W(512x16) ----
    v8f dd = {0.f,0.f,0.f,0.f,0.f,0.f,0.f,0.f};
    #pragma unroll
    for (int q = 0; q < 16; ++q) {
        const int kb = q * 32;
        v16h a = pack16(*(const v8h*)&Afin[Mrow][kb + half8],
                        *(const v8h*)&Afin[Mrow][kb + 16 + half8]);
        v16h bv = pack16(*(const v8h*)&Bt[Mrow][kb + half8],
                         *(const v8h*)&Bt[Mrow][kb + 16 + half8]);
        dd = wmma_f16(a, bv, dd);
    }

    // D layout: VGPR r -> point = r + half8, out channel = lane&15
    #pragma unroll
    for (int r = 0; r < 8; ++r) {
        const int prow = r + half8;
        const float cn = cntS[prow];
        const float res = dd[r] / fmaxf(cn, 1.f);
        const int ngr = tileBase + prow;
        out[(size_t)ngr * 16 + Mrow] = res;
    }
}

extern "C" void kernel_launch(void* const* d_in, const int* in_sizes, int n_in,
                              void* d_out, int out_size, void* d_ws, size_t ws_size,
                              hipStream_t stream) {
    (void)in_sizes; (void)n_in; (void)out_size; (void)d_ws; (void)ws_size;
    const float* input   = (const float*)d_in[0];   // [2,16,64,64,64]
    const float* pos     = (const float*)d_in[1];   // [2,8192,3]
    const float* weights = (const float*)d_in[2];   // [3,3,3,16,16]
    // d_in[3] (grid_pos) unused: grid_pos[g] == cell_index * dx exactly.
    float* out = (float*)d_out;                     // [2,8192,16]

    const int total_points = 2 * 8192;
    const int blocks = total_points / PTS;          // 1024
    hipLaunchKernelGGL(cconv_decoder_kernel, dim3(blocks), dim3(32), 0, stream,
                       input, pos, weights, out);
}
